// EdgeRegressionModel_39247411151002
// MI455X (gfx1250) — matrix operand order repaired and checked by reading, compiled, not run on hardware
//
#include <hip/hip_runtime.h>
#include <stdint.h>

typedef __attribute__((ext_vector_type(16))) __bf16 v16bf;
typedef __attribute__((ext_vector_type(2)))  __bf16 v2bf;
typedef __attribute__((ext_vector_type(8)))  float  v8f;

// ---------------------------------------------------------------- bf16 helpers
__device__ __forceinline__ unsigned short f2bf_sw(float f) {
    union { float f; uint32_t u; } x; x.f = f;
    uint32_t u = x.u;
    u += 0x7FFFu + ((u >> 16) & 1u);     // round-to-nearest-even
    return (unsigned short)(u >> 16);
}

// packed f32x2 -> bf16x2 (hardware v_cvt_pk_bf16_f32 when available)
__device__ __forceinline__ uint32_t pk_bf16(float a, float b) {
#if __has_builtin(__builtin_amdgcn_cvt_pk_bf16_f32)
    union { v2bf v; uint32_t u; } x;
    x.v = __builtin_amdgcn_cvt_pk_bf16_f32(a, b);
    return x.u;
#else
    return (uint32_t)f2bf_sw(a) | ((uint32_t)f2bf_sw(b) << 16);
#endif
}

__device__ __forceinline__ unsigned short f2bf(float a) {
#if __has_builtin(__builtin_amdgcn_cvt_pk_bf16_f32)
    return (unsigned short)(pk_bf16(a, 0.0f) & 0xFFFFu);
#else
    return f2bf_sw(a);
#endif
}

union FragAB { uint4 q[2]; v16bf v; };   // 16 bf16 = 32B = 8 VGPRs/lane

// ---------------------------------------------------------------- tiny kernels
__global__ void k_zero(float* p, int n) {
    int i = blockIdx.x * blockDim.x + threadIdx.x;
    if (i < n) p[i] = 0.0f;
}

__global__ void k_deg(const int* __restrict__ ei, float* __restrict__ deg, int E) {
    int e = blockIdx.x * blockDim.x + threadIdx.x;
    if (e < E) atomicAdd(&deg[ei[E + e]], 1.0f);   // dst = edge_index[1]
}

__global__ void k_inv(float* __restrict__ deg, int n) {
    int i = blockIdx.x * blockDim.x + threadIdx.x;
    if (i < n) deg[i] = rsqrtf(deg[i] + 1.0f);     // self-loop adds 1
}

// f32 -> bf16, 2 elements/thread, packed 32-bit stores (n must be even)
__global__ void k_cvt2(const float* __restrict__ in, uint32_t* __restrict__ out, int n2) {
    int i = blockIdx.x * blockDim.x + threadIdx.x;
    if (i < n2) {
        float2 v = *(const float2*)(in + 2 * (size_t)i);
        out[i] = pk_bf16(v.x, v.y);
    }
}

// W[K,N] f32 (row-major) -> Wt[N,K] bf16 (row-major); thread handles (n, 2k)
__global__ void k_wt(const float* __restrict__ W, uint32_t* __restrict__ Wt, int K, int N) {
    int i = blockIdx.x * blockDim.x + threadIdx.x;       // over N * K/2
    int K2 = K >> 1;
    if (i < N * K2) {
        int n = i / K2, k2 = i - n * K2;
        float a = W[(size_t)(2 * k2 + 0) * N + n];
        float b = W[(size_t)(2 * k2 + 1) * N + n];
        Wt[(size_t)n * K2 + k2] = pk_bf16(a, b);
    }
}

// ---------------------------------------------------------------- generic WMMA GEMM
// out[M,N] = act(A[M,K](bf16) @ Bt[N,K]^T(bf16) + bias)
// one wave -> 16x32 output tile (two 16x16 WMMA tiles sharing the A fragment),
// 4 waves/block along M.  K is a compile-time constant (full unroll).
template <int K, bool RELU, bool OUT_BF16>
__global__ void k_gemm(const unsigned short* __restrict__ A,
                       const unsigned short* __restrict__ Bt,
                       const float* __restrict__ bias,
                       void* __restrict__ out, int M, int N) {
    const int wave = threadIdx.x >> 5;
    const int lane = threadIdx.x & 31;
    const int mt = blockIdx.x * 4 + wave;
    if (mt * 16 >= M) return;                       // wave-uniform exit
    const int nt0 = blockIdx.y * 2;                 // two adjacent N tiles
    const int l15 = lane & 15;
    const int hi  = lane >> 4;
    const unsigned short* ap  = A  + (size_t)(mt * 16 + l15) * K + hi * 8;
    const unsigned short* bp0 = Bt + (size_t)(nt0 * 16 + l15) * K + hi * 8;
    const unsigned short* bp1 = bp0 + (size_t)16 * K;
    v8f c0 = {}, c1 = {};
#pragma unroll
    for (int k = 0; k < K; k += 32) {
        FragAB a, b0, b1;
        a.q[0]  = *(const uint4*)(ap + k);
        a.q[1]  = *(const uint4*)(ap + k + 16);
        b0.q[0] = *(const uint4*)(bp0 + k);
        b0.q[1] = *(const uint4*)(bp0 + k + 16);
        b1.q[0] = *(const uint4*)(bp1 + k);
        b1.q[1] = *(const uint4*)(bp1 + k + 16);
        c0 = __builtin_amdgcn_wmma_f32_16x16x32_bf16(false, a.v, false, b0.v,
                                                     (short)0, c0, false, false);
        c1 = __builtin_amdgcn_wmma_f32_16x16x32_bf16(false, a.v, false, b1.v,
                                                     (short)0, c1, false, false);
    }
    const int gn0 = nt0 * 16 + l15;
    const int gn1 = gn0 + 16;
    const float bi0 = bias ? bias[gn0] : 0.0f;
    const float bi1 = bias ? bias[gn1] : 0.0f;
#pragma unroll
    for (int r = 0; r < 8; ++r) {
        const int gm = mt * 16 + r + hi * 8;
        float v0 = c0[r] + bi0;
        float v1 = c1[r] + bi1;
        if (RELU) { v0 = fmaxf(v0, 0.0f); v1 = fmaxf(v1, 0.0f); }
        if (OUT_BF16) {
            ((unsigned short*)out)[(size_t)gm * N + gn0] = f2bf(v0);
            ((unsigned short*)out)[(size_t)gm * N + gn1] = f2bf(v1);
        } else {
            ((float*)out)[(size_t)gm * N + gn0] = v0;
            ((float*)out)[(size_t)gm * N + gn1] = v1;
        }
    }
}

// ---------------------------------------------------------------- GCN scatter + finalize
__global__ void k_scatter(const int* __restrict__ ei, const float* __restrict__ inv,
                          const float* __restrict__ t, float* __restrict__ agg, int E) {
    int tid = blockIdx.x * blockDim.x + threadIdx.x;
    if (tid >= E * 32) return;
    int e = tid >> 5, q = tid & 31;
    int s = ei[e], d = ei[E + e];
    float nrm = inv[s] * inv[d];
    const float4 v = *(const float4*)(t + (size_t)s * 128 + q * 4);
    float* o = agg + (size_t)d * 128 + q * 4;
    atomicAdd(o + 0, v.x * nrm);
    atomicAdd(o + 1, v.y * nrm);
    atomicAdd(o + 2, v.z * nrm);
    atomicAdd(o + 3, v.w * nrm);
}

// 2 features/thread -> packed bf16 store; n2 = N*128/2
template <bool RELU>
__global__ void k_gcn_fin(const float* __restrict__ agg, const float* __restrict__ t,
                          const float* __restrict__ inv, const float* __restrict__ b,
                          uint32_t* __restrict__ hout, int n2) {
    int i = blockIdx.x * blockDim.x + threadIdx.x;
    if (i >= n2) return;
    int node = i >> 6, f2 = i & 63;                 // 64 feature-pairs per node
    float iv = inv[node];
    float iv2 = iv * iv;
    float2 ag = *(const float2*)(agg + 2 * (size_t)i);
    float2 tt = *(const float2*)(t + 2 * (size_t)i);
    float v0 = ag.x + tt.x * iv2 + b[2 * f2 + 0];
    float v1 = ag.y + tt.y * iv2 + b[2 * f2 + 1];
    if (RELU) { v0 = fmaxf(v0, 0.0f); v1 = fmaxf(v1, 0.0f); }
    hout[i] = pk_bf16(v0, v1);
}

// ---------------------------------------------------------------- fused edge MLP
// 16 edges / block, 128 threads (4 waves): relu(relu(ea@eW1+b1)@eW2+b2) -> bf16 [E,64]
__global__ void k_edgemlp(const float* __restrict__ ea,
                          const unsigned short* __restrict__ eW1t, const float* __restrict__ eb1,
                          const unsigned short* __restrict__ eW2t, const float* __restrict__ eb2,
                          unsigned short* __restrict__ eemb, int E) {
    __shared__ unsigned short sEA[16 * 32];
    __shared__ unsigned short sH[16 * 128];
    const int e0 = blockIdx.x * 16;
    const int tid = threadIdx.x;
    // stage 0: load + convert 16x32 tile (pairs -> packed stores)
#pragma unroll
    for (int i = 0; i < 2; ++i) {
        int idx = i * 128 + tid;             // 256 pairs
        int row = idx >> 4, c2 = idx & 15;
        int e = min(e0 + row, E - 1);
        float2 v = *(const float2*)(ea + (size_t)e * 32 + 2 * c2);
        *(uint32_t*)(sEA + row * 32 + 2 * c2) = pk_bf16(v.x, v.y);
    }
    __syncthreads();
    const int wave = tid >> 5, lane = tid & 31;
    const int l15 = lane & 15, hi = lane >> 4;
    // stage 1: hidden = relu(ea @ eW1 + eb1)  [16x128], K=32
#pragma unroll
    for (int j = 0; j < 2; ++j) {
        const int nt = wave * 2 + j;
        const unsigned short* ap = sEA + l15 * 32 + hi * 8;
        const unsigned short* bp = eW1t + (size_t)(nt * 16 + l15) * 32 + hi * 8;
        FragAB a, b;
        a.q[0] = *(const uint4*)(ap);      a.q[1] = *(const uint4*)(ap + 16);
        b.q[0] = *(const uint4*)(bp);      b.q[1] = *(const uint4*)(bp + 16);
        v8f c = {};
        c = __builtin_amdgcn_wmma_f32_16x16x32_bf16(false, a.v, false, b.v,
                                                    (short)0, c, false, false);
        const int gn = nt * 16 + l15;
        const float bi = eb1[gn];
#pragma unroll
        for (int r = 0; r < 8; ++r)
            sH[(r + hi * 8) * 128 + gn] = f2bf(fmaxf(c[r] + bi, 0.0f));
    }
    __syncthreads();
    // stage 2: emb = relu(hidden @ eW2 + eb2)  [16x64], K=128
    {
        const int nt = wave;
        const unsigned short* ap = sH + l15 * 128 + hi * 8;
        const unsigned short* bp = eW2t + (size_t)(nt * 16 + l15) * 128 + hi * 8;
        v8f c = {};
#pragma unroll
        for (int k = 0; k < 128; k += 32) {
            FragAB a, b;
            a.q[0] = *(const uint4*)(ap + k);  a.q[1] = *(const uint4*)(ap + k + 16);
            b.q[0] = *(const uint4*)(bp + k);  b.q[1] = *(const uint4*)(bp + k + 16);
            c = __builtin_amdgcn_wmma_f32_16x16x32_bf16(false, a.v, false, b.v,
                                                        (short)0, c, false, false);
        }
        const int gn = nt * 16 + l15;
        const float bi = eb2[gn];
#pragma unroll
        for (int r = 0; r < 8; ++r) {
            int e = e0 + r + hi * 8;
            if (e < E) eemb[(size_t)e * 64 + gn] = f2bf(fmaxf(c[r] + bi, 0.0f));
        }
    }
}

// ---------------------------------------------------------------- fused edge head
// 16 edges / block: feats=[h[src]|h[dst]|eemb] (16x320 bf16 in LDS)
// z1=relu(feats@hW1+b1) 16x128 ; z2=relu(z1@hW2+b2) 16x64 ; out=z2@hW3+b3
__global__ void k_head(const int* __restrict__ ei,
                       const unsigned short* __restrict__ h_bf,
                       const unsigned short* __restrict__ eemb,
                       const unsigned short* __restrict__ hW1t, const float* __restrict__ hb1,
                       const unsigned short* __restrict__ hW2t, const float* __restrict__ hb2,
                       const float* __restrict__ hW3, const float* __restrict__ hb3,
                       float* __restrict__ out, int E) {
    __shared__ unsigned short feats[16 * 320];
    __shared__ unsigned short z1[16 * 128];
    __shared__ float z2[16 * 64];
    __shared__ int sS[16], sD[16];
    const int e0 = blockIdx.x * 16;
    const int tid = threadIdx.x;
    if (tid < 16) {
        int e = min(e0 + tid, E - 1);
        sS[tid] = ei[e];
        sD[tid] = ei[E + e];
    }
    __syncthreads();
    {   // gather concat features; 8 threads/row, 8-bf16 (16B) chunks
        const int row = tid >> 3, c0 = (tid & 7) * 8;
        const int s = sS[row], d = sD[row];
        const int e = min(e0 + row, E - 1);
        // cols [0,128): h[src]; [128,256): h[dst]; [256,320): eemb (edge)
        for (int c = c0; c < 320; c += 64) {
            uint4 v;
            if      (c < 128) v = *(const uint4*)(h_bf + (size_t)s * 128 + c);
            else if (c < 256) v = *(const uint4*)(h_bf + (size_t)d * 128 + (c - 128));
            else              v = *(const uint4*)(eemb + (size_t)e * 64 + (c - 256));
            *(uint4*)(feats + row * 320 + c) = v;
        }
    }
    __syncthreads();
    const int wave = tid >> 5, lane = tid & 31;
    const int l15 = lane & 15, hi = lane >> 4;
    // z1 = relu(feats @ hW1 + hb1), K=320, 8 N-tiles (2 per wave)
#pragma unroll
    for (int j = 0; j < 2; ++j) {
        const int nt = wave * 2 + j;
        const unsigned short* ap = feats + l15 * 320 + hi * 8;
        const unsigned short* bp = hW1t + (size_t)(nt * 16 + l15) * 320 + hi * 8;
        v8f c = {};
#pragma unroll
        for (int k = 0; k < 320; k += 32) {
            FragAB a, b;
            a.q[0] = *(const uint4*)(ap + k);  a.q[1] = *(const uint4*)(ap + k + 16);
            b.q[0] = *(const uint4*)(bp + k);  b.q[1] = *(const uint4*)(bp + k + 16);
            c = __builtin_amdgcn_wmma_f32_16x16x32_bf16(false, a.v, false, b.v,
                                                        (short)0, c, false, false);
        }
        const int gn = nt * 16 + l15;
        const float bi = hb1[gn];
#pragma unroll
        for (int r = 0; r < 8; ++r)
            z1[(r + hi * 8) * 128 + gn] = f2bf(fmaxf(c[r] + bi, 0.0f));
    }
    __syncthreads();
    // z2 = relu(z1 @ hW2 + hb2), K=128, 4 N-tiles (1 per wave)
    {
        const int nt = wave;
        const unsigned short* ap = z1 + l15 * 128 + hi * 8;
        const unsigned short* bp = hW2t + (size_t)(nt * 16 + l15) * 128 + hi * 8;
        v8f c = {};
#pragma unroll
        for (int k = 0; k < 128; k += 32) {
            FragAB a, b;
            a.q[0] = *(const uint4*)(ap + k);  a.q[1] = *(const uint4*)(ap + k + 16);
            b.q[0] = *(const uint4*)(bp + k);  b.q[1] = *(const uint4*)(bp + k + 16);
            c = __builtin_amdgcn_wmma_f32_16x16x32_bf16(false, a.v, false, b.v,
                                                        (short)0, c, false, false);
        }
        const int gn = nt * 16 + l15;
        const float bi = hb2[gn];
#pragma unroll
        for (int r = 0; r < 8; ++r)
            z2[(r + hi * 8) * 64 + gn] = fmaxf(c[r] + bi, 0.0f);
    }
    __syncthreads();
    // out = z2 @ hW3 + hb3  (K=64, N=1) in f32
    if (tid < 16 && (e0 + tid) < E) {
        float sum = hb3[0];
#pragma unroll
        for (int f = 0; f < 64; ++f) sum += z2[tid * 64 + f] * hW3[f];
        out[e0 + tid] = sum;
    }
}

// ---------------------------------------------------------------- launcher
extern "C" void kernel_launch(void* const* d_in, const int* in_sizes, int n_in,
                              void* d_out, int out_size, void* d_ws, size_t ws_size,
                              hipStream_t stream) {
    const float* x   = (const float*)d_in[0];
    const float* ea  = (const float*)d_in[1];
    const int*   ei  = (const int*)d_in[2];
    const float* nW1 = (const float*)d_in[3];  const float* nb1 = (const float*)d_in[4];
    const float* nW2 = (const float*)d_in[5];  const float* nb2 = (const float*)d_in[6];
    const float* eW1 = (const float*)d_in[7];  const float* eb1 = (const float*)d_in[8];
    const float* eW2 = (const float*)d_in[9];  const float* eb2 = (const float*)d_in[10];
    const float* gW0 = (const float*)d_in[11]; const float* gb0 = (const float*)d_in[12];
    const float* gW1 = (const float*)d_in[13]; const float* gb1 = (const float*)d_in[14];
    const float* gW2 = (const float*)d_in[15]; const float* gb2 = (const float*)d_in[16];
    const float* hW1 = (const float*)d_in[17]; const float* hb1 = (const float*)d_in[18];
    const float* hW2 = (const float*)d_in[19]; const float* hb2 = (const float*)d_in[20];
    const float* hW3 = (const float*)d_in[21]; const float* hb3 = (const float*)d_in[22];
    float* out = (float*)d_out;
    const int N = in_sizes[0] / 128;     // 50000
    const int E = in_sizes[2] / 2;       // 500000

    // ---- workspace carving (aligned) ----
    char* base = (char*)d_ws;
    size_t off = 0;
    auto carve = [&](size_t bytes) -> void* {
        off = (off + 255) & ~(size_t)255;
        void* p = base + off; off += bytes; return p;
    };
    float*          inv   = (float*)carve(sizeof(float) * N);
    unsigned short* x_bf  = (unsigned short*)carve(2ull * N * 128);
    unsigned short* nW1t  = (unsigned short*)carve(2ull * 128 * 128);
    unsigned short* nW2t  = (unsigned short*)carve(2ull * 128 * 128);
    unsigned short* eW1t  = (unsigned short*)carve(2ull * 128 * 32);
    unsigned short* eW2t  = (unsigned short*)carve(2ull * 64 * 128);
    unsigned short* g0t   = (unsigned short*)carve(2ull * 128 * 128);
    unsigned short* g1t   = (unsigned short*)carve(2ull * 128 * 128);
    unsigned short* g2t   = (unsigned short*)carve(2ull * 128 * 128);
    unsigned short* hW1t  = (unsigned short*)carve(2ull * 128 * 320);
    unsigned short* hW2t  = (unsigned short*)carve(2ull * 64 * 128);
    unsigned short* bufA  = (unsigned short*)carve(2ull * N * 128);   // node hidden
    unsigned short* hbf0  = (unsigned short*)carve(2ull * N * 128);   // ping
    unsigned short* hbf1  = (unsigned short*)carve(2ull * N * 128);   // pong
    unsigned short* eemb  = (unsigned short*)carve(2ull * E * 64);
    float*          t     = (float*)carve(sizeof(float) * (size_t)N * 128);
    float*          agg   = (float*)carve(sizeof(float) * (size_t)N * 128);

    const int T = 256;
    auto nb = [](long long n, int t) { return (unsigned)((n + t - 1) / t); };

    // degrees -> inv sqrt (with self loop)
    k_zero<<<nb(N, T), T, 0, stream>>>(inv, N);
    k_deg <<<nb(E, T), T, 0, stream>>>(ei, inv, E);
    k_inv <<<nb(N, T), T, 0, stream>>>(inv, N);

    // conversions (packed)
    k_cvt2<<<nb((long long)N * 64, T), T, 0, stream>>>(x, (uint32_t*)x_bf, N * 64);
    k_wt<<<nb(128 * 64, T),  T, 0, stream>>>(nW1, (uint32_t*)nW1t, 128, 128);
    k_wt<<<nb(128 * 64, T),  T, 0, stream>>>(nW2, (uint32_t*)nW2t, 128, 128);
    k_wt<<<nb(128 * 16, T),  T, 0, stream>>>(eW1, (uint32_t*)eW1t, 32, 128);
    k_wt<<<nb(64 * 64, T),   T, 0, stream>>>(eW2, (uint32_t*)eW2t, 128, 64);
    k_wt<<<nb(128 * 64, T),  T, 0, stream>>>(gW0, (uint32_t*)g0t, 128, 128);
    k_wt<<<nb(128 * 64, T),  T, 0, stream>>>(gW1, (uint32_t*)g1t, 128, 128);
    k_wt<<<nb(128 * 64, T),  T, 0, stream>>>(gW2, (uint32_t*)g2t, 128, 128);
    k_wt<<<nb(128 * 160, T), T, 0, stream>>>(hW1, (uint32_t*)hW1t, 320, 128);
    k_wt<<<nb(64 * 64, T),   T, 0, stream>>>(hW2, (uint32_t*)hW2t, 128, 64);

    // node MLP: x -> node_emb (bf16 in hbf0)
    const int mtN = (N + 15) / 16;
    dim3 gN((mtN + 3) / 4, 128 / 32), blk(128);   // 16x32 per wave -> grid.y = N/32
    k_gemm<128, true, true><<<gN, blk, 0, stream>>>(x_bf, nW1t, nb1, bufA, N, 128);
    k_gemm<128, true, true><<<gN, blk, 0, stream>>>(bufA, nW2t, nb2, hbf0, N, 128);

    // edge MLP fused: edge_attr -> eemb (bf16 [E,64])
    k_edgemlp<<<nb(E, 16), blk, 0, stream>>>(ea, eW1t, eb1, eW2t, eb2, eemb, E);

    // 3 GCN layers
    const long long NF  = (long long)N * 128;
    const long long NF2 = NF / 2;
    // layer 0: hbf0 -> hbf1 (relu)
    k_gemm<128, false, false><<<gN, blk, 0, stream>>>(hbf0, g0t, (const float*)nullptr, t, N, 128);
    k_zero<<<nb(NF, T), T, 0, stream>>>(agg, (int)NF);
    k_scatter<<<nb((long long)E * 32, T), T, 0, stream>>>(ei, inv, t, agg, E);
    k_gcn_fin<true><<<nb(NF2, T), T, 0, stream>>>(agg, t, inv, gb0, (uint32_t*)hbf1, (int)NF2);
    // layer 1: hbf1 -> hbf0 (relu)
    k_gemm<128, false, false><<<gN, blk, 0, stream>>>(hbf1, g1t, (const float*)nullptr, t, N, 128);
    k_zero<<<nb(NF, T), T, 0, stream>>>(agg, (int)NF);
    k_scatter<<<nb((long long)E * 32, T), T, 0, stream>>>(ei, inv, t, agg, E);
    k_gcn_fin<true><<<nb(NF2, T), T, 0, stream>>>(agg, t, inv, gb1, (uint32_t*)hbf0, (int)NF2);
    // layer 2: hbf0 -> hbf1 (no relu)
    k_gemm<128, false, false><<<gN, blk, 0, stream>>>(hbf0, g2t, (const float*)nullptr, t, N, 128);
    k_zero<<<nb(NF, T), T, 0, stream>>>(agg, (int)NF);
    k_scatter<<<nb((long long)E * 32, T), T, 0, stream>>>(ei, inv, t, agg, E);
    k_gcn_fin<false><<<nb(NF2, T), T, 0, stream>>>(agg, t, inv, gb2, (uint32_t*)hbf1, (int)NF2);

    // fused edge head -> out [E]
    k_head<<<nb(E, 16), blk, 0, stream>>>(ei, hbf1, eemb, hW1t, hb1, hW2t, hb2,
                                          hW3, hb3, out, E);
}
